// MultiPNAAutoEncoder_85899345979
// MI455X (gfx1250) — compile-verified, matrix operand order checked by reading, not run on hardware
//
#include <hip/hip_runtime.h>

// MultiPNAAutoEncoder forward for gfx1250 (MI455X), wave32 + WMMA bf16.
// Workspace requirement: ~870 MB (harness-provided d_ws assumed large enough).
//
// WMMA fragment feeding (ISA 7.12.2, 16-bit operands):
//  - A: lane(lm,half) needs row (m16+lm), K runs [k0+8h,+8) and [k0+16+8h,+8)
//       -> two b128 LDS loads + shuffle (compiler hoists all k-steps to regs).
//  - B: pre-swizzled at weight-pack time into [kstep][ntile][lane][16] order
//       -> one contiguous 32-byte global load per fragment (L2-hot weights).
// Epilogues use a per-block full-tile fast path so the dominant path has no
// per-row exec-mask guards around the aggregation atomics.

#define TPB 256

typedef __attribute__((ext_vector_type(16))) __bf16         v16bf;
typedef __attribute__((ext_vector_type(16))) unsigned short v16u;
typedef __attribute__((ext_vector_type(8)))  unsigned short v8u;
typedef __attribute__((ext_vector_type(8)))  float          v8f;

__device__ __forceinline__ unsigned short f2bf(float f) {
  unsigned int u = __builtin_bit_cast(unsigned int, f);
  u += 0x7FFFu + ((u >> 16) & 1u);          // round-to-nearest-even
  return (unsigned short)(u >> 16);
}
// order-preserving float<->uint key for integer atomicMin/Max
__device__ __forceinline__ unsigned int fkey(float f) {
  unsigned int u = __builtin_bit_cast(unsigned int, f);
  return (u & 0x80000000u) ? ~u : (u | 0x80000000u);
}
__device__ __forceinline__ float funkey(unsigned int k) {
  unsigned int u = (k & 0x80000000u) ? (k & 0x7FFFFFFFu) : ~k;
  return __builtin_bit_cast(float, u);
}
// ISA 7.12.2: 16-bit A/B fragment element i of lane-half -> K index inside 32-step
__device__ __forceinline__ int frag_k(int i, int half) {
  int v = i >> 1, b = i & 1;
  return ((v < 4) ? 2 * v : 8 + 2 * v) + b + 8 * half;
}
// A fragment: two contiguous 8-elem runs from a row-major bf16 row in LDS
__device__ __forceinline__ v16u load_a_frag(const unsigned short* rowp, int k0,
                                            int half) {
  v8u lo = *(const v8u*)(rowp + k0 + 8 * half);
  v8u hi = *(const v8u*)(rowp + k0 + 16 + 8 * half);
  return __builtin_shufflevector(lo, hi, 0, 1, 2, 3, 4, 5, 6, 7, 8, 9, 10, 11,
                                 12, 13, 14, 15);
}

// ---------------------------------------------------------------------------
// Generic WMMA GEMM: C[M,ldc] = act(A_bf16[M,KP] @ Bswz[KP,NP] + bias[Nl])
// Block = 256 threads = 8 waves; each wave owns a 16-row tile of a 128-row
// block slab staged in LDS. B is fragment-swizzled packed weights in L2.
// ---------------------------------------------------------------------------
template <int KP, int NP>
__global__ __launch_bounds__(TPB) void wmma_gemm(
    const unsigned short* __restrict__ A, const unsigned short* __restrict__ Bp,
    const float* __restrict__ bias, float* __restrict__ C, int M, int Nl,
    int ldc, int relu) {
  __shared__ __align__(16) unsigned short As[128 * KP];
  const int tid = threadIdx.x;
  const int rowBase = blockIdx.x * 128;
  const bool full = (rowBase + 128 <= M);
  constexpr int CH = KP / 8;  // 8-element chunks per row
  for (int i = tid; i < 128 * CH; i += TPB) {
    int r = i / CH, c8 = i % CH;
    int gr = rowBase + r;
    v8u v = {};
    if (full || gr < M) v = *(const v8u*)(A + (long)gr * KP + c8 * 8);
    *(v8u*)(As + r * KP + c8 * 8) = v;
  }
  __syncthreads();
  const int wave = tid >> 5, lane = tid & 31;
  const int lm = lane & 15, half = lane >> 4;
  const unsigned short* arow = As + (wave * 16 + lm) * KP;
  for (int n0 = 0; n0 < NP; n0 += 16) {
    v8f acc = {};
#pragma unroll
    for (int k0 = 0; k0 < KP; k0 += 32) {
      v16u au = load_a_frag(arow, k0, half);
      v16u bu = *(const v16u*)(Bp + ((k0 >> 5) * (NP / 16) + (n0 >> 4)) * 512 +
                               lane * 16);
      acc = __builtin_amdgcn_wmma_f32_16x16x32_bf16(
          false, __builtin_bit_cast(v16bf, au), false,
          __builtin_bit_cast(v16bf, bu), (short)0, acc, false, false);
    }
    int col = n0 + lm;
    if (col < Nl) {
      float bv = bias[col];
      if (full) {
#pragma unroll
        for (int r = 0; r < 8; ++r) {
          int gr = rowBase + wave * 16 + r + 8 * half;
          float v = acc[r] + bv;
          if (relu) v = fmaxf(v, 0.f);
          C[(long)gr * ldc + col] = v;
        }
      } else {
#pragma unroll
        for (int r = 0; r < 8; ++r) {
          int gr = rowBase + wave * 16 + r + 8 * half;
          if (gr < M) {
            float v = acc[r] + bv;
            if (relu) v = fmaxf(v, 0.f);
            C[(long)gr * ldc + col] = v;
          }
        }
      }
    }
  }
}

// ---------------------------------------------------------------------------
// PNA edge kernel: m = [nx[dst] | nx[src] | ea] @ Wfused[128,208] + bfused,
// reduced directly into per-(node,channel) sum/sumsq/min/max via atomics.
// Wfused is fragment-swizzled: [4 ksteps][13 ntiles][32 lanes][16].
// ---------------------------------------------------------------------------
__global__ __launch_bounds__(TPB) void pna_edge_wmma(
    const unsigned short* __restrict__ nxbf /*[N,64]*/,
    const unsigned short* __restrict__ eabf /*[E,64]*/,
    const int* __restrict__ src, const int* __restrict__ dst,
    const unsigned short* __restrict__ Wf, const float* __restrict__ bf,
    float* __restrict__ aggsum, float* __restrict__ aggsqr,
    unsigned int* __restrict__ aggmin, unsigned int* __restrict__ aggmax,
    int E_) {
  __shared__ __align__(16) unsigned short As[128 * 128];  // 32 KB of 320/WGP
  __shared__ int sSrc[128];
  __shared__ int sDst[128];
  const int tid = threadIdx.x;
  const int e0 = blockIdx.x * 128;
  const bool full = (e0 + 128 <= E_);
  if (tid == 0) __builtin_prefetch(Wf, 0, 1);  // global_prefetch_b8
  for (int i = tid; i < 128; i += TPB) {
    int e = e0 + i;
    sSrc[i] = (full || e < E_) ? src[e] : 0;
    sDst[i] = (full || e < E_) ? dst[e] : 0;
  }
  __syncthreads();
  // gather rows as 8-channel (16B) chunks: [0,5)=dst, [5,10)=src, [10,15)=ea
  for (int i = tid; i < 128 * 16; i += TPB) {
    int r = i >> 4, c8 = i & 15;
    v8u v = {};
    if (full || e0 + r < E_) {
      if (c8 < 5)
        v = *(const v8u*)(nxbf + (long)sDst[r] * 64 + c8 * 8);
      else if (c8 < 10)
        v = *(const v8u*)(nxbf + (long)sSrc[r] * 64 + (c8 - 5) * 8);
      else if (c8 < 15)
        v = *(const v8u*)(eabf + (long)(e0 + r) * 64 + (c8 - 10) * 8);
    }
    *(v8u*)(As + r * 128 + c8 * 8) = v;
  }
  __syncthreads();
  const int wave = tid >> 5, lane = tid & 31;
  const int lm = lane & 15, half = lane >> 4;
  const unsigned short* arow = As + (wave * 16 + lm) * 128;
  for (int n0 = 0; n0 < 208; n0 += 16) {
    v8f acc = {};
#pragma unroll
    for (int k0 = 0; k0 < 128; k0 += 32) {
      v16u au = load_a_frag(arow, k0, half);
      v16u bu =
          *(const v16u*)(Wf + ((k0 >> 5) * 13 + (n0 >> 4)) * 512 + lane * 16);
      acc = __builtin_amdgcn_wmma_f32_16x16x32_bf16(
          false, __builtin_bit_cast(v16bf, au), false,
          __builtin_bit_cast(v16bf, bu), (short)0, acc, false, false);
    }
    int col = n0 + lm;
    if (col < 200) {
      float bias = bf[col];
      if (full) {
#pragma unroll
        for (int r = 0; r < 8; ++r) {
          int row = wave * 16 + r + 8 * half;
          float v = acc[r] + bias;
          long idx = (long)sDst[row] * 200 + col;
          atomicAdd(&aggsum[idx], v);
          atomicAdd(&aggsqr[idx], v * v);
          atomicMin(&aggmin[idx], fkey(v));
          atomicMax(&aggmax[idx], fkey(v));
        }
      } else {
#pragma unroll
        for (int r = 0; r < 8; ++r) {
          int row = wave * 16 + r + 8 * half;
          if (e0 + row < E_) {
            float v = acc[r] + bias;
            long idx = (long)sDst[row] * 200 + col;
            atomicAdd(&aggsum[idx], v);
            atomicAdd(&aggsqr[idx], v * v);
            atomicMin(&aggmin[idx], fkey(v));
            atomicMax(&aggmax[idx], fkey(v));
          }
        }
      }
    }
  }
}

// ---------------------------------------------------------------------------
// Small helper kernels
// ---------------------------------------------------------------------------
__global__ void k_deg(const int* __restrict__ dst, float* __restrict__ deg,
                      long E_) {
  long e = (long)blockIdx.x * blockDim.x + threadIdx.x;
  if (e < E_) atomicAdd(&deg[dst[e]], 1.0f);
}

__global__ void k_avglog(const float* __restrict__ deg,
                         float* __restrict__ out, long n_) {
  __shared__ float red[TPB];
  float s = 0.f;
  for (long i = (long)blockIdx.x * blockDim.x + threadIdx.x; i < n_;
       i += (long)gridDim.x * blockDim.x)
    s += logf(deg[i] + 1.0f);
  red[threadIdx.x] = s;
  __syncthreads();
  for (int st = TPB / 2; st > 0; st >>= 1) {
    if (threadIdx.x < st) red[threadIdx.x] += red[threadIdx.x + st];
    __syncthreads();
  }
  if (threadIdx.x == 0) atomicAdd(out, red[0]);
}

// f32 [rows,cols] -> bf16 [rows,KP] padded
__global__ void cvt_pad_bf(const float* __restrict__ s,
                           unsigned short* __restrict__ d, long rows, int cols,
                           int KP) {
  long i = (long)blockIdx.x * blockDim.x + threadIdx.x;
  if (i >= rows * (long)KP) return;
  long r = i / KP;
  int c = (int)(i % KP);
  d[i] = (c < cols) ? f2bf(s[r * cols + c]) : (unsigned short)0;
}

// f32 weight [K,Nl] -> fragment-swizzled bf16 [(KP/32)*(NP/16)][32][16]
__global__ void pack_w_frag(const float* __restrict__ W,
                            unsigned short* __restrict__ out, int K, int Nl,
                            int KP, int NP) {
  int total = (KP / 32) * (NP / 16) * 512;
  int idx = blockIdx.x * blockDim.x + threadIdx.x;
  if (idx >= total) return;
  int fid = idx >> 9, lane = (idx >> 4) & 31, i = idx & 15;
  int ntiles = NP / 16;
  int s = fid / ntiles, nt = fid % ntiles;
  int half = lane >> 4, lm = lane & 15;
  int k = s * 32 + frag_k(i, half);
  int n = nt * 16 + lm;
  out[idx] = (k < K && n < Nl) ? f2bf(W[k * Nl + n]) : (unsigned short)0;
}

// Fragment-swizzled Wfused: rows 0:80 = preW[:80]; rows 80:120 = eW@preW[80:120]
__global__ void build_wfused(const float* __restrict__ preW,
                             const float* __restrict__ eW,
                             unsigned short* __restrict__ Wf) {
  int idx = blockIdx.x * blockDim.x + threadIdx.x;
  if (idx >= 4 * 13 * 512) return;
  int fid = idx >> 9, lane = (idx >> 4) & 31, i = idx & 15;
  int s = fid / 13, nt = fid % 13;
  int half = lane >> 4, lm = lane & 15;
  int row = s * 32 + frag_k(i, half);
  int col = nt * 16 + lm;
  float v = 0.f;
  if (col < 200 && row < 120) {
    int t = col / 40, f = col % 40;
    if (row < 80)
      v = preW[(t * 120 + row) * 40 + f];
    else {
      int r = row - 80;
      float a = 0.f;
      for (int q = 0; q < 40; ++q)
        a += eW[r * 40 + q] * preW[(t * 120 + 80 + q) * 40 + f];
      v = a;
    }
  }
  Wf[idx] = f2bf(v);
}

__global__ void build_bfused(const float* __restrict__ preW,
                             const float* __restrict__ preB,
                             const float* __restrict__ eb,
                             float* __restrict__ bf) {
  int col = blockIdx.x * blockDim.x + threadIdx.x;
  if (col >= 208) return;
  float v = 0.f;
  if (col < 200) {
    int t = col / 40, f = col % 40;
    v = preB[t * 40 + f];
    for (int q = 0; q < 40; ++q)
      v += eb[q] * preW[(t * 120 + 80 + q) * 40 + f];
  }
  bf[col] = v;
}

// in-place: sum->mean, sqr->std, min/max keys -> decoded floats (0 if deg==0)
__global__ void finalize_agg(float* __restrict__ s, float* __restrict__ sq,
                             unsigned int* __restrict__ mnu,
                             unsigned int* __restrict__ mxu,
                             const float* __restrict__ deg, long n_) {
  long i = (long)blockIdx.x * blockDim.x + threadIdx.x;
  if (i >= n_ * 200) return;
  long node = i / 200;
  float d = deg[node], cnt = fmaxf(d, 1.f);
  float mean = s[i] / cnt, m2 = sq[i] / cnt;
  s[i] = mean;
  sq[i] = sqrtf(fmaxf(m2 - mean * mean, 0.f) + 1e-5f);
  ((float*)mnu)[i] = (d > 0.f) ? funkey(mnu[i]) : 0.f;
  ((float*)mxu)[i] = (d > 0.f) ? funkey(mxu[i]) : 0.f;
}

// per-node tower MLP with degree scalers folded into the weight read
__global__ void pna_post(const float* __restrict__ nx,
                         const float* __restrict__ mean,
                         const float* __restrict__ stdv,
                         const float* __restrict__ mnf,
                         const float* __restrict__ mxf,
                         const float* __restrict__ deg,
                         const float* __restrict__ avglog,
                         const float* __restrict__ postW,
                         const float* __restrict__ postB,
                         float* __restrict__ hpost, long n_) {
  long i = (long)blockIdx.x * blockDim.x + threadIdx.x;
  if (i >= n_ * 40) return;
  long node = i / 40;
  int to = (int)(i % 40), t = to / 8, o = to % 8;
  float avg = avglog[0] / (float)n_;
  float ld = logf(fmaxf(deg[node], 1.f) + 1.f);
  float s1 = ld / avg, s2 = avg / ld;
  const float* W = postW + (long)t * 520 * 8;
  float acc = postB[t * 8 + o];
  const float* xv = nx + node * 40;
  for (int k = 0; k < 40; ++k) acc += xv[k] * W[k * 8 + o];
  long base = node * 200 + t * 40;
  for (int j = 0; j < 40; ++j) {
    float a0 = mean[base + j], a1 = mnf[base + j], a2 = mxf[base + j],
          a3 = stdv[base + j];
    int r0 = 40 + j, r1 = 80 + j, r2 = 120 + j, r3 = 160 + j;
    acc += a0 * (W[r0 * 8 + o] + s1 * W[(r0 + 160) * 8 + o] + s2 * W[(r0 + 320) * 8 + o]);
    acc += a1 * (W[r1 * 8 + o] + s1 * W[(r1 + 160) * 8 + o] + s2 * W[(r1 + 320) * 8 + o]);
    acc += a2 * (W[r2 * 8 + o] + s1 * W[(r2 + 160) * 8 + o] + s2 * W[(r2 + 320) * 8 + o]);
    acc += a3 * (W[r3 * 8 + o] + s1 * W[(r3 + 160) * 8 + o] + s2 * W[(r3 + 320) * 8 + o]);
  }
  hpost[i] = acc;
}

// batch-norm stats via LDS float atomics then 2*C global atomics per block
__global__ void bn_stats(const float* __restrict__ h, float* __restrict__ stats,
                         long R, int C) {
  __shared__ float ls[128];
  for (int i = threadIdx.x; i < 2 * C; i += blockDim.x) ls[i] = 0.f;
  __syncthreads();
  long total = R * (long)C;
  for (long i = (long)blockIdx.x * blockDim.x + threadIdx.x; i < total;
       i += (long)gridDim.x * blockDim.x) {
    int c = (int)(i % C);
    float v = h[i];
    atomicAdd(&ls[c], v);
    atomicAdd(&ls[C + c], v * v);
  }
  __syncthreads();
  for (int i = threadIdx.x; i < 2 * C; i += blockDim.x)
    atomicAdd(&stats[i], ls[i]);
}

// nx = (nx + relu(bn(hlin)))/2 ; maintain padded bf16 mirror
__global__ void conv_update(float* __restrict__ nx,
                            unsigned short* __restrict__ nxbf,
                            const float* __restrict__ h,
                            const float* __restrict__ stats,
                            const float* __restrict__ g,
                            const float* __restrict__ b, long n_) {
  long i = (long)blockIdx.x * blockDim.x + threadIdx.x;
  if (i >= n_ * 40) return;
  int c = (int)(i % 40);
  float mu = stats[c] / (float)n_;
  float var = stats[40 + c] / (float)n_ - mu * mu;
  float hv = g[c] * (h[i] - mu) * rsqrtf(var + 1e-5f) + b[c];
  float v = (nx[i] + fmaxf(hv, 0.f)) * 0.5f;
  nx[i] = v;
  nxbf[(i / 40) * 64 + c] = f2bf(v);
}

__global__ void bn_apply(const float* __restrict__ hin, float* __restrict__ hout,
                         const float* __restrict__ stats,
                         const float* __restrict__ g,
                         const float* __restrict__ b, long R, int C) {
  long i = (long)blockIdx.x * blockDim.x + threadIdx.x;
  if (i >= R * (long)C) return;
  int c = (int)(i % C);
  float mu = stats[c] / (float)R;
  float var = stats[C + c] / (float)R - mu * mu;
  hout[i] = g[c] * (hin[i] - mu) * rsqrtf(var + 1e-5f) + b[c];
}

__global__ void enc_kernel(const float* __restrict__ nx,
                           const float* __restrict__ W,
                           const float* __restrict__ b, float* __restrict__ z,
                           long n_) {
  long i = (long)blockIdx.x * blockDim.x + threadIdx.x;
  if (i >= n_ * 12) return;
  long n = i / 12;
  int o = (int)(i % 12);
  float a = b[o];
  const float* xv = nx + n * 40;
  for (int k = 0; k < 40; ++k) a += xv[k] * W[k * 12 + o];
  z[i] = a;
}

__global__ void feat_build(const float* __restrict__ z,
                           const int* __restrict__ src,
                           const int* __restrict__ dst,
                           unsigned short* __restrict__ fb, long E_) {
  long i = (long)blockIdx.x * blockDim.x + threadIdx.x;
  if (i >= E_ * 32) return;
  long e = i / 32;
  int c = (int)(i % 32);
  float v = 0.f;
  if (c < 12)
    v = z[(long)src[e] * 12 + c];
  else if (c < 24)
    v = z[(long)dst[e] * 12 + (c - 12)];
  fb[i] = f2bf(v);
}

__global__ void err_kernel(const float* __restrict__ recon,
                           const float* __restrict__ orig,
                           float* __restrict__ err, long E_) {
  long e = (long)blockIdx.x * blockDim.x + threadIdx.x;
  if (e >= E_) return;
  float s = 0.f;
  for (int k = 0; k < 16; ++k) {
    float d = recon[e * 16 + k] - orig[e * 16 + k];
    s += d * d;
  }
  err[e] = s * (1.f / 16.f);
}

__global__ void cls_kernel(const float* __restrict__ z,
                           const int* __restrict__ src,
                           const int* __restrict__ dst,
                           const float* __restrict__ orig,
                           const float* __restrict__ err,
                           const float* __restrict__ W1, const float* __restrict__ b1,
                           const float* __restrict__ W2, const float* __restrict__ b2,
                           const float* __restrict__ W3, const float* __restrict__ b3,
                           float* __restrict__ out, long E_) {
  long e = (long)blockIdx.x * blockDim.x + threadIdx.x;
  if (e >= E_) return;
  float f[41];
  const float* zs = z + (long)src[e] * 12;
  const float* zd = z + (long)dst[e] * 12;
  for (int k = 0; k < 12; ++k) f[k] = zs[k];
  for (int k = 0; k < 12; ++k) f[12 + k] = zd[k];
  for (int k = 0; k < 16; ++k) f[24 + k] = orig[e * 16 + k];
  f[40] = err[e];
  float h1[50];
  for (int j = 0; j < 50; ++j) {
    float a = b1[j];
    for (int k = 0; k < 41; ++k) a += f[k] * W1[k * 50 + j];
    h1[j] = fmaxf(a, 0.f);
  }
  float h2[25];
  for (int j = 0; j < 25; ++j) {
    float a = b2[j];
    for (int k = 0; k < 50; ++k) a += h1[k] * W2[k * 25 + j];
    h2[j] = fmaxf(a, 0.f);
  }
  for (int o = 0; o < 2; ++o) {
    float a = b3[o];
    for (int k = 0; k < 25; ++k) a += h2[k] * W3[k * 2 + o];
    out[e * 2 + o] = a;
  }
}

// ---------------------------------------------------------------------------
extern "C" void kernel_launch(void* const* d_in, const int* in_sizes, int n_in,
                              void* d_out, int out_size, void* d_ws,
                              size_t ws_size, hipStream_t stream) {
  (void)n_in; (void)out_size; (void)ws_size;
  const long N = in_sizes[0] / 32;  // 100000
  const long E = in_sizes[1] / 2;   // 500000

  // --- input pointer table (setup_inputs insertion-order flattening) ---
  const float* x     = (const float*)d_in[0];
  const int*   eidx  = (const int*)d_in[1];
  const float* eattr = (const float*)d_in[2];
  const int *src = eidx, *dst = eidx + E;
  int p = 3;
  const float* nodeW = (const float*)d_in[p++]; const float* nodeB = (const float*)d_in[p++];
  const float* edgeW = (const float*)d_in[p++]; const float* edgeB = (const float*)d_in[p++];
  const float *cEW[2], *cEB[2], *cPreW[2], *cPreB[2], *cPostW[2], *cPostB[2], *cLinW[2], *cLinB[2];
  for (int l = 0; l < 2; ++l) {
    cEW[l] = (const float*)d_in[p++]; cEB[l] = (const float*)d_in[p++];
    cPreW[l] = (const float*)d_in[p++]; cPreB[l] = (const float*)d_in[p++];
    cPostW[l] = (const float*)d_in[p++]; cPostB[l] = (const float*)d_in[p++];
    cLinW[l] = (const float*)d_in[p++]; cLinB[l] = (const float*)d_in[p++];
  }
  const float *bnG[2], *bnB[2];
  for (int l = 0; l < 2; ++l) { bnG[l] = (const float*)d_in[p++]; bnB[l] = (const float*)d_in[p++]; }
  const float* encW = (const float*)d_in[p++]; const float* encB = (const float*)d_in[p++];
  const float* deW = (const float*)d_in[p++]; const float* deB = (const float*)d_in[p++];
  const float *dlW[2], *dlB[2], *dlG[2], *dlBB[2];
  for (int l = 0; l < 2; ++l) {
    dlW[l] = (const float*)d_in[p++]; dlB[l] = (const float*)d_in[p++];
    dlG[l] = (const float*)d_in[p++]; dlBB[l] = (const float*)d_in[p++];
  }
  const float* dfW = (const float*)d_in[p++]; const float* dfB = (const float*)d_in[p++];
  const float* W1 = (const float*)d_in[p++]; const float* b1 = (const float*)d_in[p++];
  const float* W2 = (const float*)d_in[p++]; const float* b2 = (const float*)d_in[p++];
  const float* W3 = (const float*)d_in[p++]; const float* b3 = (const float*)d_in[p++];
  float* out = (float*)d_out;

  // --- workspace carve-up ---
  char* ws = (char*)d_ws;
  size_t off = 0;
  auto carve = [&](size_t bytes) { off = (off + 255) & ~(size_t)255; size_t o = off; off += bytes; return ws + o; };
  float* nx       = (float*)carve(N * 40 * 4);
  unsigned short* nxbf   = (unsigned short*)carve(N * 64 * 2);
  float* ea       = (float*)carve(E * 40 * 4);
  unsigned short* eabf   = (unsigned short*)carve(E * 64 * 2);
  unsigned short* xbf    = (unsigned short*)carve(N * 32 * 2);
  unsigned short* eainbf = (unsigned short*)carve(E * 32 * 2);
  float* deg      = (float*)carve(N * 4);
  float* avglog   = (float*)carve(256);
  float* aggsum   = (float*)carve(N * 200 * 4);
  float* aggsqr   = (float*)carve(N * 200 * 4);
  unsigned int* aggmin = (unsigned int*)carve(N * 200 * 4);
  unsigned int* aggmax = (unsigned int*)carve(N * 200 * 4);
  float* hpost    = (float*)carve(N * 40 * 4);
  unsigned short* hpostbf = (unsigned short*)carve(N * 64 * 2);
  float* hlin     = (float*)carve(N * 40 * 4);
  float* bnstats  = (float*)carve(128 * 4);
  unsigned short* wfused = (unsigned short*)carve(4 * 13 * 512 * 2);
  float* bfused   = (float*)carve(208 * 4);
  float* z        = (float*)carve(N * 12 * 4);
  unsigned short* featbf = (unsigned short*)carve(E * 32 * 2);
  float* hdec     = (float*)carve(E * 40 * 4);
  float* hdecB    = (float*)carve(E * 40 * 4);
  unsigned short* hdecbf = (unsigned short*)carve(E * 64 * 2);
  float* recon    = (float*)carve(E * 16 * 4);
  float* err      = (float*)carve(E * 4);
  unsigned short* pk = (unsigned short*)carve(2 * 3 * 512 * 2);  // weight-pack buf

  auto cdiv = [](long a, long b) { return (int)((a + b - 1) / b); };

  // --- degree + avg_log ---
  hipMemsetAsync(deg, 0, N * 4, stream);
  hipMemsetAsync(avglog, 0, 256, stream);
  k_deg<<<cdiv(E, TPB), TPB, 0, stream>>>(dst, deg, E);
  k_avglog<<<1024, TPB, 0, stream>>>(deg, avglog, N);

  // --- embeddings (WMMA) ---
  cvt_pad_bf<<<cdiv(N * 32, TPB), TPB, 0, stream>>>(x, xbf, N, 32, 32);
  cvt_pad_bf<<<cdiv(E * 32, TPB), TPB, 0, stream>>>(eattr, eainbf, E, 16, 32);
  pack_w_frag<<<cdiv(1 * 3 * 512, TPB), TPB, 0, stream>>>(nodeW, pk, 32, 40, 32, 48);
  wmma_gemm<32, 48><<<cdiv(N, 128), TPB, 0, stream>>>(xbf, pk, nodeB, nx, (int)N, 40, 40, 0);
  pack_w_frag<<<cdiv(1 * 3 * 512, TPB), TPB, 0, stream>>>(edgeW, pk, 16, 40, 32, 48);
  wmma_gemm<32, 48><<<cdiv(E, 128), TPB, 0, stream>>>(eainbf, pk, edgeB, ea, (int)E, 40, 40, 0);
  cvt_pad_bf<<<cdiv(N * 64, TPB), TPB, 0, stream>>>(nx, nxbf, N, 40, 64);
  cvt_pad_bf<<<cdiv(E * 64, TPB), TPB, 0, stream>>>(ea, eabf, E, 40, 64);

  // --- PNA conv layers ---
  for (int l = 0; l < 2; ++l) {
    build_wfused<<<cdiv(4 * 13 * 512, TPB), TPB, 0, stream>>>(cPreW[l], cEW[l], wfused);
    build_bfused<<<1, TPB, 0, stream>>>(cPreW[l], cPreB[l], cEB[l], bfused);
    hipMemsetAsync(aggsum, 0, N * 200 * 4, stream);
    hipMemsetAsync(aggsqr, 0, N * 200 * 4, stream);
    hipMemsetAsync(aggmin, 0xFF, N * 200 * 4, stream);  // +inf key
    hipMemsetAsync(aggmax, 0x00, N * 200 * 4, stream);  // -inf key
    pna_edge_wmma<<<cdiv(E, 128), TPB, 0, stream>>>(nxbf, eabf, src, dst, wfused,
                                                    bfused, aggsum, aggsqr,
                                                    aggmin, aggmax, (int)E);
    finalize_agg<<<cdiv(N * 200, TPB), TPB, 0, stream>>>(aggsum, aggsqr, aggmin, aggmax, deg, N);
    pna_post<<<cdiv(N * 40, TPB), TPB, 0, stream>>>(
        nx, aggsum, aggsqr, (const float*)aggmin, (const float*)aggmax, deg,
        avglog, cPostW[l], cPostB[l], hpost, N);
    cvt_pad_bf<<<cdiv(N * 64, TPB), TPB, 0, stream>>>(hpost, hpostbf, N, 40, 64);
    pack_w_frag<<<cdiv(2 * 3 * 512, TPB), TPB, 0, stream>>>(cLinW[l], pk, 40, 40, 64, 48);
    wmma_gemm<64, 48><<<cdiv(N, 128), TPB, 0, stream>>>(hpostbf, pk, cLinB[l], hlin, (int)N, 40, 40, 0);
    hipMemsetAsync(bnstats, 0, 128 * 4, stream);
    bn_stats<<<2048, TPB, 0, stream>>>(hlin, bnstats, N, 40);
    conv_update<<<cdiv(N * 40, TPB), TPB, 0, stream>>>(nx, nxbf, hlin, bnstats, bnG[l], bnB[l], N);
  }

  // --- encoder + decoder ---
  enc_kernel<<<cdiv(N * 12, TPB), TPB, 0, stream>>>(nx, encW, encB, z, N);
  feat_build<<<cdiv(E * 32, TPB), TPB, 0, stream>>>(z, src, dst, featbf, E);
  pack_w_frag<<<cdiv(1 * 3 * 512, TPB), TPB, 0, stream>>>(deW, pk, 24, 40, 32, 48);
  wmma_gemm<32, 48><<<cdiv(E, 128), TPB, 0, stream>>>(featbf, pk, deB, hdec, (int)E, 40, 40, 1);
  for (int l = 0; l < 2; ++l) {
    cvt_pad_bf<<<cdiv(E * 64, TPB), TPB, 0, stream>>>(hdec, hdecbf, E, 40, 64);
    pack_w_frag<<<cdiv(2 * 3 * 512, TPB), TPB, 0, stream>>>(dlW[l], pk, 40, 40, 64, 48);
    wmma_gemm<64, 48><<<cdiv(E, 128), TPB, 0, stream>>>(hdecbf, pk, dlB[l], hdecB, (int)E, 40, 40, 1);
    hipMemsetAsync(bnstats, 0, 128 * 4, stream);
    bn_stats<<<2048, TPB, 0, stream>>>(hdecB, bnstats, E, 40);
    bn_apply<<<cdiv(E * 40, TPB), TPB, 0, stream>>>(hdecB, hdec, bnstats, dlG[l], dlBB[l], E, 40);
  }
  cvt_pad_bf<<<cdiv(E * 64, TPB), TPB, 0, stream>>>(hdec, hdecbf, E, 40, 64);
  pack_w_frag<<<cdiv(2 * 1 * 512, TPB), TPB, 0, stream>>>(dfW, pk, 40, 16, 64, 16);
  wmma_gemm<64, 16><<<cdiv(E, 128), TPB, 0, stream>>>(hdecbf, pk, dfB, recon, (int)E, 16, 16, 0);
  err_kernel<<<cdiv(E, TPB), TPB, 0, stream>>>(recon, eattr, err, E);

  // --- classifier ---
  cls_kernel<<<cdiv(E, TPB), TPB, 0, stream>>>(z, src, dst, eattr, err, W1, b1,
                                               W2, b2, W3, b3, out, E);
}
